// SetGatherReadout_12429635354781
// MI455X (gfx1250) — compile-verified, matrix operand order checked by reading, not run on hardware
//
#include <hip/hip_runtime.h>
#include <math.h>
#include <stdint.h>

#define D 128
#define STEPS 8
#define SC 64   // nodes staged to LDS per chunk (64 rows x 512 B = 32 KB)

typedef __bf16 bf16;
typedef float v8f __attribute__((ext_vector_type(8)));
typedef bf16 v16bf __attribute__((ext_vector_type(16)));

__device__ __forceinline__ unsigned short f2bf(float f) {
  union { bf16 b; unsigned short u; } cv;
  cv.b = (bf16)f;
  return cv.u;
}

__device__ __forceinline__ float sigmoidf_(float x) {
  return 1.0f / (1.0f + __expf(-x));
}

// ---------------------------------------------------------------------------
// W [256,512] f32 -> Wt [512,256] bf16 (transposed so each lane's B fragment
// is two contiguous 16B loads).
// ---------------------------------------------------------------------------
__global__ void wt_convert_kernel(const float* __restrict__ W,
                                  unsigned short* __restrict__ Wt) {
  int idx = blockIdx.x * blockDim.x + threadIdx.x;
  const int total = (2 * D) * (4 * D);
  if (idx >= total) return;
  int k = idx / (4 * D);
  int n = idx % (4 * D);
  Wt[(size_t)n * (2 * D) + k] = f2bf(W[idx]);
}

// ---------------------------------------------------------------------------
// graph_indicator is sorted: starts[g] = first node of graph g, starts[G] = N.
// ---------------------------------------------------------------------------
__global__ void seg_bounds_kernel(const int* __restrict__ gi,
                                  int* __restrict__ starts, int N, int G) {
  int n = blockIdx.x * blockDim.x + threadIdx.x;
  if (n >= N) return;
  int cur = gi[n];
  if (n == 0) {
    for (int g = 0; g <= cur; ++g) starts[g] = 0;
  } else {
    int prev = gi[n - 1];
    for (int g = prev + 1; g <= cur; ++g) starts[g] = n;
  }
  if (n == N - 1) {
    for (int g = cur + 1; g <= G; ++g) starts[g] = N;
  }
}

// ---------------------------------------------------------------------------
// memory_state = 0, carry half of E (f32 view == d_out, plus bf16 copy) = 0.
// ---------------------------------------------------------------------------
__global__ void init_state_kernel(float* __restrict__ Ef,
                                  unsigned short* __restrict__ Ebf,
                                  float* __restrict__ mem, int G) {
  int idx = blockIdx.x * blockDim.x + threadIdx.x;
  if (idx >= G * D) return;
  int g = idx / D, d = idx % D;
  mem[idx] = 0.0f;
  Ef[(size_t)g * 2 * D + d] = 0.0f;
  Ebf[(size_t)g * 2 * D + d] = f2bf(0.0f);
}

// ---------------------------------------------------------------------------
// Fused per-graph attention: stage a 64-node chunk of x into LDS with
// global_load_async_to_lds_b128 (one HBM read per row, no VGPR round-trip),
// then score + online softmax + weighted readout entirely from LDS.
// One workgroup (128 thr = 4 waves) per graph; node ranges contiguous.
// ---------------------------------------------------------------------------
__global__ __launch_bounds__(128) void attn_readout_kernel(
    const float* __restrict__ x, const int* __restrict__ starts,
    float* __restrict__ Ef, unsigned short* __restrict__ Ebf, int G) {
  __shared__ float xs[SC * D];  // 32 KB staged node rows
  __shared__ float sc[SC];
  __shared__ float red[128];
  const int g = blockIdx.x;
  const int tid = threadIdx.x;
  const int wave = tid >> 5, lane = tid & 31;
  const int s0 = starts[g], s1 = starts[g + 1];

  // carry fragment: each wave's 32 lanes cover all 128 dims (float4/lane)
  const float4 cf = ((const float4*)(Ef + (size_t)g * 2 * D))[lane];

  float acc = 0.0f;   // readout accumulator for dim `tid`
  float m = -1e30f;   // running max
  float ssum = 0.0f;  // running exp-sum

  for (int base = s0; base < s1; base += SC) {
    int cnt = min(SC, s1 - base);

    // phase 0: async-stage rows into LDS (lane strides 16B across one row)
    for (int j = wave; j < cnt; j += 4) {
      unsigned long long ga =
          (unsigned long long)(uintptr_t)(x + (size_t)(base + j) * D) +
          (unsigned)(lane * 16);
      unsigned la = (unsigned)(uintptr_t)(&xs[j * D]) + (unsigned)(lane * 16);
      asm volatile("global_load_async_to_lds_b128 %0, %1, off"
                   :
                   : "v"(la), "v"(ga)
                   : "memory");
    }
    asm volatile("s_wait_asynccnt 0x0" ::: "memory");
    __syncthreads();

    // phase 1: scores (one node per wave-iteration, wave-level dot reduce)
    for (int j = wave; j < cnt; j += 4) {
      const float4 xv = ((const float4*)(xs + j * D))[lane];
      float d = xv.x * cf.x + xv.y * cf.y + xv.z * cf.z + xv.w * cf.w;
#pragma unroll
      for (int off = 16; off > 0; off >>= 1) d += __shfl_xor(d, off, 32);
      if (lane == 0) sc[j] = d;
    }
    __syncthreads();

    // phase 2: chunk max -> new running max, rescale running state
    float lm = -1e30f;
    for (int j = tid; j < cnt; j += 128) lm = fmaxf(lm, sc[j]);
    red[tid] = lm;
    __syncthreads();
    for (int s = 64; s > 0; s >>= 1) {
      if (tid < s) red[tid] = fmaxf(red[tid], red[tid + s]);
      __syncthreads();
    }
    float m_new = fmaxf(m, red[0]);
    __syncthreads();
    float scale = __expf(m - m_new);
    acc *= scale;
    ssum *= scale;
    m = m_new;

    // phase 3: weights in-place, chunk exp-sum
    float wsum = 0.0f;
    for (int j = tid; j < cnt; j += 128) {
      float w = __expf(sc[j] - m_new);
      sc[j] = w;
      wsum += w;
    }
    red[tid] = wsum;
    __syncthreads();
    for (int s = 64; s > 0; s >>= 1) {
      if (tid < s) red[tid] += red[tid + s];
      __syncthreads();
    }
    ssum += red[0];
    __syncthreads();

    // phase 4: weighted accumulate from LDS (dim `tid` per thread)
    for (int j = 0; j < cnt; ++j) {
      acc += sc[j] * xs[j * D + tid];
    }
    __syncthreads();
  }

  float r = (ssum > 0.0f) ? acc / ssum : 0.0f;  // empty graph -> 0 readout
  Ef[(size_t)g * 2 * D + D + tid] = r;
  Ebf[(size_t)g * 2 * D + D + tid] = f2bf(r);
}

// ---------------------------------------------------------------------------
// Fused gates GEMM + LSTM update: z = E(bf16) @ W(bf16) via
// v_wmma_f32_16x16x32_bf16, with the gate nonlinearity applied in the
// epilogue. Tile remap: wave w computes column tile t*8+w for gate t, so
// acc[0..3] = (u,f,c,o) for the SAME (row, dim) in every thread -> no z
// buffer, no second kernel, ~100 MB/step of HBM traffic removed.
// A-fragment per ISA layout: lane<16 row=lane, K {0..7,16..23}; lane>=16
// row=lane-16, K {8..15,24..31} -> two 16B LDS reads. B from transposed
// Wt[512][256] -> two 16B global reads (Wt = 256 KB, L2-resident).
// ---------------------------------------------------------------------------
__global__ __launch_bounds__(256) void gates_lstm_wmma_kernel(
    const unsigned short* __restrict__ E, const unsigned short* __restrict__ Wt,
    const float* __restrict__ bias, float* __restrict__ mem,
    float* __restrict__ Ef, unsigned short* __restrict__ Ebf, int G) {
  __shared__ uint4 eQ[512];  // 16 rows x 256 bf16 = 8 KB (32 uint4 per row)
  const int tid = threadIdx.x;
  const int wave = tid >> 5, lane = tid & 31;
  const int g0 = blockIdx.x * 16;

  for (int i = tid; i < 512; i += 256) {
    int r = i >> 5, qc = i & 31;
    uint4 v = make_uint4(0u, 0u, 0u, 0u);
    int gg = g0 + r;
    if (gg < G) v = ((const uint4*)(E + (size_t)gg * 2 * D))[qc];
    eQ[i] = v;
  }
  __syncthreads();

  const int half = lane >> 4;  // 0: K-offsets {0..7,16..23}; 1: {8..15,24..31}
  const int lr = lane & 15;

  union U { uint4 q[2]; v16bf v; };
  v8f acc[4];
  const v8f vzero = {0.f, 0.f, 0.f, 0.f, 0.f, 0.f, 0.f, 0.f};
#pragma unroll
  for (int t = 0; t < 4; ++t) acc[t] = vzero;

#pragma unroll
  for (int kt = 0; kt < 8; ++kt) {
    U a;
    a.q[0] = eQ[lr * 32 + kt * 4 + half];
    a.q[1] = eQ[lr * 32 + kt * 4 + half + 2];
#pragma unroll
    for (int t = 0; t < 4; ++t) {
      int nt = t * 8 + wave;  // gate t, column block `wave` within the gate
      const uint4* wq =
          (const uint4*)(Wt + (size_t)(nt * 16 + lr) * (2 * D) + kt * 32);
      U b;
      b.q[0] = wq[half];
      b.q[1] = wq[half + 2];
      acc[t] = __builtin_amdgcn_wmma_f32_16x16x32_bf16(
          false, a.v, false, b.v, (short)0, acc[t], false, false);
    }
  }

  // Epilogue: C/D layout lane<16 -> N=lane, M=vgpr; lane>=16 -> N=lane-16,
  // M=vgpr+8. Thread holds u,f,c,o for dim dcol = wave*16 + N, rows M.
  const int dcol = wave * 16 + lr;
  const float bu = bias[dcol];
  const float bf_ = bias[D + dcol];
  const float bc = bias[2 * D + dcol];
  const float bo = bias[3 * D + dcol];
#pragma unroll
  for (int v = 0; v < 8; ++v) {
    int gg = g0 + half * 8 + v;
    if (gg < G) {
      float u = acc[0][v] + bu;
      float f = acc[1][v] + bf_;
      float c = acc[2][v] + bc;
      float o = acc[3][v] + bo;
      float mv = mem[(size_t)gg * D + dcol];
      mv = sigmoidf_(f) * mv + sigmoidf_(u) * tanhf(c);
      mem[(size_t)gg * D + dcol] = mv;
      float carry = sigmoidf_(o) * tanhf(mv);
      Ef[(size_t)gg * 2 * D + dcol] = carry;
      Ebf[(size_t)gg * 2 * D + dcol] = f2bf(carry);
    }
  }
}

// ---------------------------------------------------------------------------
extern "C" void kernel_launch(void* const* d_in, const int* in_sizes, int n_in,
                              void* d_out, int out_size, void* d_ws,
                              size_t ws_size, hipStream_t stream) {
  const float* x = (const float*)d_in[0];     // [N,128] f32
  const int* gi = (const int*)d_in[1];        // [N] i32 (sorted)
  const float* W = (const float*)d_in[2];     // [256,512] f32
  const float* bias = (const float*)d_in[3];  // [512] f32
  const int N = in_sizes[1];
  const int G = out_size / (2 * D);

  float* Ef = (float*)d_out;  // E = [carry | readout], f32 view IS the output

  char* ws = (char*)d_ws;
  size_t off = 0;
  auto a256 = [](size_t v) { return (v + 255) & ~(size_t)255; };
  int* starts = (int*)(ws + off);
  off = a256(off + (size_t)(G + 1) * sizeof(int));
  unsigned short* Ebf = (unsigned short*)(ws + off);
  off = a256(off + (size_t)G * 2 * D * sizeof(unsigned short));
  unsigned short* Wt = (unsigned short*)(ws + off);
  off = a256(off + (size_t)(4 * D) * (2 * D) * sizeof(unsigned short));
  float* mem = (float*)(ws + off);
  off = a256(off + (size_t)G * D * sizeof(float));
  (void)ws_size;
  (void)n_in;

  wt_convert_kernel<<<((2 * D) * (4 * D) + 255) / 256, 256, 0, stream>>>(W, Wt);
  seg_bounds_kernel<<<(N + 255) / 256, 256, 0, stream>>>(gi, starts, N, G);
  init_state_kernel<<<(G * D + 255) / 256, 256, 0, stream>>>(Ef, Ebf, mem, G);

  for (int step = 0; step < STEPS; ++step) {
    attn_readout_kernel<<<G, 128, 0, stream>>>(x, starts, Ef, Ebf, G);
    if (step < STEPS - 1) {  // final step's LSTM update doesn't affect output
      gates_lstm_wmma_kernel<<<(G + 15) / 16, 256, 0, stream>>>(
          Ebf, Wt, bias, mem, Ef, Ebf, G);
    }
  }
}
